// MultiHeadAttention_68994354643599
// MI455X (gfx1250) — matrix operand order repaired and checked by reading, not compiled, not run on hardware
//
#include <hip/hip_runtime.h>
#include <hip/hip_bf16.h>

typedef _Float16 f16;
typedef __attribute__((ext_vector_type(16))) _Float16 v16h;
typedef __attribute__((ext_vector_type(8)))  _Float16 v8h;
typedef __attribute__((ext_vector_type(4)))  _Float16 v4h;
typedef __attribute__((ext_vector_type(8)))  float    v8f;
typedef __attribute__((ext_vector_type(4)))  float    v4f;

#define D_MODEL 1024
#define HEADS   16
#define D_K     64
#define BATCH   2
#define SEQ     2048
#define KEEP    1843   // int(2048 * 0.9)

__device__ __forceinline__ v8f wmma_f16(v16h a, v16h b, v8f c) {
  // D = A(16x32 f16) * B(32x16 f16) + C(16x16 f32)
  return __builtin_amdgcn_wmma_f32_16x16x32_f16(
      false, a, false, b, (short)0, c, false, false);
}

// reduce across the 16 lanes that hold one matrix row (wave32, half-wave groups)
__device__ __forceinline__ float rmax16(float v) {
#pragma unroll
  for (int m = 8; m > 0; m >>= 1) v = fmaxf(v, __shfl_xor(v, m, 16));
  return v;
}
__device__ __forceinline__ float rsum16(float v) {
#pragma unroll
  for (int m = 8; m > 0; m >>= 1) v += __shfl_xor(v, m, 16);
  return v;
}

__device__ __forceinline__ v4h load4_cvt(const float* p) {
  v4f t = *(const v4f*)p;
  v4h r;
  r[0] = (f16)t[0]; r[1] = (f16)t[1]; r[2] = (f16)t[2]; r[3] = (f16)t[3];
  return r;
}
__device__ __forceinline__ v4h load4_cvt(const f16* p) {
  return *(const v4h*)p;
}

// ---------------------------------------------------------------------------
// C[m,n] = sum_k A[m,k] * W[n,k] + bias[n]    (x @ W^T + b)
// Block 256 thr = 8 waves; tile 128x128x32; wave = 32x64 = 2x4 WMMA tiles.
// LDT=40 halves (80B rows): 16B-aligned fragment chunks -> ds_load_b128.
// ---------------------------------------------------------------------------
template <typename TA, typename TO>
__global__ __launch_bounds__(256)
void gemm_xwt_bias(const TA* __restrict__ A, const float* __restrict__ W,
                   const float* __restrict__ bias, TO* __restrict__ C,
                   int M, int N, int K) {
  constexpr int BM = 128, BN = 128, BK = 32, LDT = 40;
  __shared__ __align__(16) f16 As[BM * LDT];
  __shared__ __align__(16) f16 Bs[BN * LDT];
  const int tid = threadIdx.x;
  const int lane = tid & 31;
  const int wave = tid >> 5;
  const int wm = wave & 3;   // 4 row-blocks of 32
  const int wn = wave >> 2;  // 2 col-blocks of 64
  const int m0 = blockIdx.y * BM;
  const int n0 = blockIdx.x * BN;

  v8f zero = {};
  v8f acc[2][4];
#pragma unroll
  for (int i = 0; i < 2; ++i)
#pragma unroll
    for (int j = 0; j < 4; ++j) acc[i][j] = zero;

  for (int k0 = 0; k0 < K; k0 += BK) {
    // vectorized staging: 4 elems per chunk (global 16B/8B load, ds_store_b64)
    for (int i = tid * 4; i < BM * BK; i += 256 * 4) {
      int r = i >> 5, c = i & 31;
      *(v4h*)&As[r * LDT + c] = load4_cvt(&A[(size_t)(m0 + r) * K + k0 + c]);
      *(v4h*)&Bs[r * LDT + c] = load4_cvt(&W[(size_t)(n0 + r) * K + k0 + c]);
    }
    if (k0 + BK < K) {  // hint next tiles into cache (global_prefetch_b8)
      __builtin_prefetch(&A[(size_t)(m0 + (tid >> 1)) * K + k0 + BK], 0, 1);
      __builtin_prefetch(&W[(size_t)(n0 + (tid >> 1)) * K + k0 + BK], 0, 1);
    }
    __syncthreads();

    v16h af[2], bf[4];
#pragma unroll
    for (int i = 0; i < 2; ++i) {
      int row = wm * 32 + i * 16 + (lane & 15);
      int k8 = (lane >> 4) * 16;
#pragma unroll
      for (int j = 0; j < 8; ++j) {
        af[i][j]     = As[row * LDT + k8 + j];
        af[i][8 + j] = As[row * LDT + k8 + 8 + j];
      }
    }
#pragma unroll
    for (int j = 0; j < 4; ++j) {
      int col = wn * 64 + j * 16 + (lane & 15);
      int kk = (lane >> 4) * 16;
#pragma unroll
      for (int t = 0; t < 16; ++t) bf[j][t] = Bs[col * LDT + kk + t];
    }
#pragma unroll
    for (int i = 0; i < 2; ++i)
#pragma unroll
      for (int j = 0; j < 4; ++j) acc[i][j] = wmma_f16(af[i], bf[j], acc[i][j]);
    __syncthreads();
  }

#pragma unroll
  for (int i = 0; i < 2; ++i)
#pragma unroll
    for (int j = 0; j < 4; ++j) {
      int gc = n0 + wn * 64 + j * 16 + (lane & 15);
      float bb = bias[gc];
#pragma unroll
      for (int v = 0; v < 8; ++v) {
        int gr = m0 + wm * 32 + i * 16 + (lane >> 4) * 8 + v;
        C[(size_t)gr * N + gc] = (TO)(acc[i][j][v] + bb);
      }
    }
}

__global__ void zero_f32(float* p, int n) {
  int i = blockIdx.x * blockDim.x + threadIdx.x;
  if (i < n) p[i] = 0.0f;
}

// ---------------------------------------------------------------------------
// Pass 1: per (b,h, 64-query block): sweep1 = online row max/denominator
// (one butterfly pair per row per 64-col k-block), sweep2 = recompute probs
// scaled by rcp(l) and atomically accumulate column sums.
// ---------------------------------------------------------------------------
__global__ __launch_bounds__(128)
void attn_colsum(const f16* __restrict__ Qh, const f16* __restrict__ Kh,
                 float* __restrict__ colsum) {
  constexpr int LQ = 72;  // 144B rows: 16B aligned
  __shared__ __align__(16) f16 Qs[64 * LQ];
  __shared__ __align__(16) f16 Ks[64 * LQ];
  const int tid = threadIdx.x, lane = tid & 31, w = tid >> 5;
  const int bh = blockIdx.x, b = bh >> 4, h = bh & 15;
  const int q0 = blockIdx.y * 64;
  const size_t qbase = ((size_t)b * SEQ + q0) * D_MODEL + (size_t)h * D_K;
  const size_t kbase = ((size_t)b * SEQ) * D_MODEL + (size_t)h * D_K;

  for (int i = tid * 8; i < 64 * 64; i += 128 * 8) {
    int r = i >> 6, c = i & 63;
    *(v8h*)&Qs[r * LQ + c] = *(const v8h*)&Qh[qbase + (size_t)r * D_MODEL + c];
  }
  __syncthreads();

  v16h aq[2];  // Q 16x64 as two 16x32 A fragments (d contraction)
#pragma unroll
  for (int cc = 0; cc < 2; ++cc) {
    int row = w * 16 + (lane & 15);
    int k8 = (lane >> 4) * 16;
#pragma unroll
    for (int j = 0; j < 8; ++j) {
      aq[cc][j]     = Qs[row * LQ + cc * 32 + k8 + j];
      aq[cc][8 + j] = Qs[row * LQ + cc * 32 + k8 + 8 + j];
    }
  }

  float m[8], l[8];
#pragma unroll
  for (int v = 0; v < 8; ++v) { m[v] = -1e30f; l[v] = 0.0f; }
  const float scale = 0.125f;  // 1/sqrt(64)

  // ---- sweep 1: row max + denominator ----
  for (int kb = 0; kb < SEQ; kb += 64) {
    for (int i = tid * 8; i < 64 * 64; i += 128 * 8) {
      int r = i >> 6, c = i & 63;
      *(v8h*)&Ks[r * LQ + c] =
          *(const v8h*)&Kh[kbase + (size_t)(kb + r) * D_MODEL + c];
    }
    __syncthreads();
    v8f st[4];
#pragma unroll
    for (int j = 0; j < 4; ++j) {
      v16h bk0, bk1;
      int col = j * 16 + (lane & 15);
      int kk = (lane >> 4) * 16;
#pragma unroll
      for (int t = 0; t < 16; ++t) {
        bk0[t] = Ks[col * LQ + kk + t];        // B[d][n] = K[n][d], d<32
        bk1[t] = Ks[col * LQ + 32 + kk + t];   // d in [32,64)
      }
      v8f s = {};
      s = wmma_f16(aq[0], bk0, s);
      s = wmma_f16(aq[1], bk1, s);
      st[j] = s;
    }
#pragma unroll
    for (int v = 0; v < 8; ++v) {
      float lm = fmaxf(fmaxf(st[0][v], st[1][v]), fmaxf(st[2][v], st[3][v]));
      float nm = fmaxf(m[v], rmax16(lm * scale));
      float corr = __expf(m[v] - nm);
      float es = 0.0f;
#pragma unroll
      for (int j = 0; j < 4; ++j) es += __expf(st[j][v] * scale - nm);
      l[v] = l[v] * corr + rsum16(es);
      m[v] = nm;
    }
    __syncthreads();
  }
  float linv[8];
#pragma unroll
  for (int v = 0; v < 8; ++v) linv[v] = __builtin_amdgcn_rcpf(l[v]);

  // ---- sweep 2: column-sum accumulation ----
  for (int kb = 0; kb < SEQ; kb += 64) {
    for (int i = tid * 8; i < 64 * 64; i += 128 * 8) {
      int r = i >> 6, c = i & 63;
      *(v8h*)&Ks[r * LQ + c] =
          *(const v8h*)&Kh[kbase + (size_t)(kb + r) * D_MODEL + c];
    }
    __syncthreads();
#pragma unroll
    for (int j = 0; j < 4; ++j) {
      v16h bk0, bk1;
      int col = j * 16 + (lane & 15);
      int kk = (lane >> 4) * 16;
#pragma unroll
      for (int t = 0; t < 16; ++t) {
        bk0[t] = Ks[col * LQ + kk + t];
        bk1[t] = Ks[col * LQ + 32 + kk + t];
      }
      v8f s = {};
      s = wmma_f16(aq[0], bk0, s);
      s = wmma_f16(aq[1], bk1, s);
      float cssum = 0.0f;
#pragma unroll
      for (int v = 0; v < 8; ++v)
        cssum += __expf(s[v] * scale - m[v]) * linv[v];
      cssum += __shfl_xor(cssum, 16, 32);  // add the two row-halves
      if (lane < 16)
        atomicAdd(&colsum[(size_t)bh * SEQ + kb + j * 16 + lane], cssum);
    }
    __syncthreads();
  }
}

// ---------------------------------------------------------------------------
// Threshold = value of the keep-th largest column sum, via binary search.
// ---------------------------------------------------------------------------
__global__ __launch_bounds__(256)
void topk_mask(const float* __restrict__ colsum, f16* __restrict__ mask) {
  __shared__ float cs[SEQ];
  __shared__ float fred[256];
  __shared__ int ired[256];
  const int tid = threadIdx.x;
  const int bh = blockIdx.x;
  const float* src = colsum + (size_t)bh * SEQ;
  for (int i = tid; i < SEQ; i += 256) cs[i] = src[i];
  __syncthreads();
  float mx = 0.0f;
  for (int i = tid; i < SEQ; i += 256) mx = fmaxf(mx, cs[i]);
  fred[tid] = mx;
  __syncthreads();
  for (int s = 128; s > 0; s >>= 1) {
    if (tid < s) fred[tid] = fmaxf(fred[tid], fred[tid + s]);
    __syncthreads();
  }
  float hi = fred[0] + 1e-6f, lo = 0.0f;
  __syncthreads();
  for (int it = 0; it < 26; ++it) {
    float t = 0.5f * (lo + hi);
    int c = 0;
    for (int i = tid; i < SEQ; i += 256) c += (cs[i] >= t) ? 1 : 0;
    ired[tid] = c;
    __syncthreads();
    for (int s = 128; s > 0; s >>= 1) {
      if (tid < s) ired[tid] += ired[tid + s];
      __syncthreads();
    }
    int total = ired[0];
    __syncthreads();
    if (total > KEEP) lo = t; else hi = t;
  }
  float thr = hi;
  for (int i = tid; i < SEQ; i += 256)
    mask[(size_t)bh * SEQ + i] = (f16)((cs[i] >= thr) ? 1.0f : 0.0f);
}

// ---------------------------------------------------------------------------
// Pass 2: flash attention with 64-column k-blocks; mask multiplies the
// numerator only (reference prunes post-softmax without renormalizing).
// ---------------------------------------------------------------------------
__global__ __launch_bounds__(128)
void attn_pv_masked(const f16* __restrict__ Qh, const f16* __restrict__ Kh,
                    const f16* __restrict__ Vh, const f16* __restrict__ mask,
                    f16* __restrict__ Out) {
  constexpr int LQ = 72;
  __shared__ __align__(16) f16 Qs[64 * LQ];
  __shared__ __align__(16) f16 Ks[64 * LQ];
  __shared__ __align__(16) f16 Vs[64 * LQ];
  __shared__ __align__(16) f16 Ps[4 * 16 * LQ];  // per-wave P 16x64 scratch
  __shared__ f16 Ms[64];
  const int tid = threadIdx.x, lane = tid & 31, w = tid >> 5;
  const int bh = blockIdx.x, b = bh >> 4, h = bh & 15;
  const int q0 = blockIdx.y * 64;
  const size_t qbase = ((size_t)b * SEQ + q0) * D_MODEL + (size_t)h * D_K;
  const size_t kvbase = ((size_t)b * SEQ) * D_MODEL + (size_t)h * D_K;

  for (int i = tid * 8; i < 64 * 64; i += 128 * 8) {
    int r = i >> 6, c = i & 63;
    *(v8h*)&Qs[r * LQ + c] = *(const v8h*)&Qh[qbase + (size_t)r * D_MODEL + c];
  }
  __syncthreads();

  v16h aq[2];
#pragma unroll
  for (int cc = 0; cc < 2; ++cc) {
    int row = w * 16 + (lane & 15);
    int k8 = (lane >> 4) * 16;
#pragma unroll
    for (int j = 0; j < 8; ++j) {
      aq[cc][j]     = Qs[row * LQ + cc * 32 + k8 + j];
      aq[cc][8 + j] = Qs[row * LQ + cc * 32 + k8 + 8 + j];
    }
  }

  v8f zero = {};
  v8f acc[4];  // out 16x64 = 4 d-tiles
#pragma unroll
  for (int t = 0; t < 4; ++t) acc[t] = zero;
  float m[8], l[8];
#pragma unroll
  for (int v = 0; v < 8; ++v) { m[v] = -1e30f; l[v] = 0.0f; }

  f16* Pw = Ps + w * 16 * LQ;
  const float scale = 0.125f;

  for (int kb = 0; kb < SEQ; kb += 64) {
    for (int i = tid * 8; i < 64 * 64; i += 128 * 8) {
      int r = i >> 6, c = i & 63;
      *(v8h*)&Ks[r * LQ + c] =
          *(const v8h*)&Kh[kvbase + (size_t)(kb + r) * D_MODEL + c];
      *(v8h*)&Vs[r * LQ + c] =
          *(const v8h*)&Vh[kvbase + (size_t)(kb + r) * D_MODEL + c];
    }
    if (tid < 64) Ms[tid] = mask[(size_t)bh * SEQ + kb + tid];
    __syncthreads();

    // S = Q K^T for 64 columns: four 16x16 tiles
    v8f st[4];
#pragma unroll
    for (int j = 0; j < 4; ++j) {
      v16h bk0, bk1;
      int col = j * 16 + (lane & 15);
      int kk = (lane >> 4) * 16;
#pragma unroll
      for (int t = 0; t < 16; ++t) {
        bk0[t] = Ks[col * LQ + kk + t];
        bk1[t] = Ks[col * LQ + 32 + kk + t];
      }
      v8f s = {};
      s = wmma_f16(aq[0], bk0, s);
      s = wmma_f16(aq[1], bk1, s);
      st[j] = s;
    }

    // online softmax update (one butterfly pair per row per k-block)
    float corr[8];
#pragma unroll
    for (int v = 0; v < 8; ++v) {
      float lm = fmaxf(fmaxf(st[0][v], st[1][v]), fmaxf(st[2][v], st[3][v]));
      float nm = fmaxf(m[v], rmax16(lm * scale));
      corr[v] = __expf(m[v] - nm);
      m[v] = nm;
    }
#pragma unroll
    for (int t = 0; t < 4; ++t)
#pragma unroll
      for (int v = 0; v < 8; ++v) acc[t][v] *= corr[v];

    int prow = (lane >> 4) * 8;
    float es[8];
#pragma unroll
    for (int v = 0; v < 8; ++v) es[v] = 0.0f;
#pragma unroll
    for (int j = 0; j < 4; ++j) {
      float mv = (float)Ms[j * 16 + (lane & 15)];
#pragma unroll
      for (int v = 0; v < 8; ++v) {
        float p = __expf(st[j][v] * scale - m[v]);
        es[v] += p;                              // denominator unmasked
        Pw[(prow + v) * LQ + j * 16 + (lane & 15)] = (f16)(p * mv);
      }
    }
#pragma unroll
    for (int v = 0; v < 8; ++v) l[v] = l[v] * corr[v] + rsum16(es[v]);

    asm volatile("s_wait_dscnt 0" ::: "memory");  // per-wave LDS RAW fence

    // P (16x64) as two A fragments; V as B fragments; PV accumulate
    v16h ap[2];
#pragma unroll
    for (int cc = 0; cc < 2; ++cc) {
      int row = lane & 15;
      int k8 = (lane >> 4) * 16;
#pragma unroll
      for (int j2 = 0; j2 < 8; ++j2) {
        ap[cc][j2]     = Pw[row * LQ + cc * 32 + k8 + j2];
        ap[cc][8 + j2] = Pw[row * LQ + cc * 32 + k8 + 8 + j2];
      }
    }
#pragma unroll
    for (int t = 0; t < 4; ++t) {
      v16h bv0, bv1;
      int col = t * 16 + (lane & 15);
      int kk = (lane >> 4) * 16;
#pragma unroll
      for (int i2 = 0; i2 < 16; ++i2) {
        bv0[i2] = Vs[(kk + i2) * LQ + col];
        bv1[i2] = Vs[(32 + kk + i2) * LQ + col];
      }
      acc[t] = wmma_f16(ap[0], bv0, acc[t]);
      acc[t] = wmma_f16(ap[1], bv1, acc[t]);
    }
    __syncthreads();
  }

  float linv[8];
#pragma unroll
  for (int v = 0; v < 8; ++v) linv[v] = __builtin_amdgcn_rcpf(l[v]);
#pragma unroll
  for (int t = 0; t < 4; ++t) {
    int d = t * 16 + (lane & 15);
#pragma unroll
    for (int v = 0; v < 8; ++v) {
      int gr = q0 + w * 16 + (lane >> 4) * 8 + v;
      Out[((size_t)b * SEQ + gr) * D_MODEL + (size_t)h * D_K + d] =
          (f16)(acc[t][v] * linv[v]);
    }
  }
}

extern "C" void kernel_launch(void* const* d_in, const int* in_sizes, int n_in,
                              void* d_out, int out_size, void* d_ws,
                              size_t ws_size, hipStream_t stream) {
  (void)in_sizes; (void)n_in; (void)out_size; (void)ws_size;
  const float* q  = (const float*)d_in[0];
  const float* k  = (const float*)d_in[1];
  const float* v  = (const float*)d_in[2];
  const float* Wq = (const float*)d_in[3];
  const float* bq = (const float*)d_in[4];
  const float* Wk = (const float*)d_in[5];
  const float* bk = (const float*)d_in[6];
  const float* Wv = (const float*)d_in[7];
  const float* bv = (const float*)d_in[8];
  const float* Wo = (const float*)d_in[9];
  const float* bo = (const float*)d_in[10];
  float* out = (float*)d_out;

  char* ws = (char*)d_ws;
  const size_t M = (size_t)BATCH * SEQ;          // 4096
  const size_t szH = M * D_MODEL * sizeof(f16);  // 8 MB
  f16* Qh = (f16*)(ws);
  f16* Kh = (f16*)(ws + szH);
  f16* Vh = (f16*)(ws + 2 * szH);
  f16* AO = (f16*)(ws + 3 * szH);
  float* colsum = (float*)(ws + 4 * szH);
  f16* mask =
      (f16*)(ws + 4 * szH + (size_t)BATCH * HEADS * SEQ * sizeof(float));

  dim3 gG(D_MODEL / 128, (int)(M / 128));  // 8 x 32
  gemm_xwt_bias<float, f16><<<gG, 256, 0, stream>>>(q, Wq, bq, Qh, (int)M, D_MODEL, D_MODEL);
  gemm_xwt_bias<float, f16><<<gG, 256, 0, stream>>>(k, Wk, bk, Kh, (int)M, D_MODEL, D_MODEL);
  gemm_xwt_bias<float, f16><<<gG, 256, 0, stream>>>(v, Wv, bv, Vh, (int)M, D_MODEL, D_MODEL);

  int csn = BATCH * HEADS * SEQ;
  zero_f32<<<(csn + 255) / 256, 256, 0, stream>>>(colsum, csn);

  dim3 gA(BATCH * HEADS, SEQ / 64);  // 32 x 32
  attn_colsum<<<gA, 128, 0, stream>>>(Qh, Kh, colsum);
  topk_mask<<<BATCH * HEADS, 256, 0, stream>>>(colsum, mask);
  attn_pv_masked<<<gA, 128, 0, stream>>>(Qh, Kh, Vh, mask, AO);

  gemm_xwt_bias<f16, float><<<gG, 256, 0, stream>>>(AO, Wo, bo, out, (int)M, D_MODEL, D_MODEL);
}
